// StableAttention_618475290727
// MI455X (gfx1250) — matrix operand order repaired and checked by reading, not compiled, not run on hardware
//
#include <hip/hip_runtime.h>
#include <hip/hip_bf16.h>

typedef __attribute__((ext_vector_type(16))) _Float16 v16h;
typedef __attribute__((ext_vector_type(8)))  _Float16 v8h;
typedef __attribute__((ext_vector_type(8)))  float    v8f;

#define DIM     1024
#define NHEADS  16
#define HD      64
#define BATCH   2
#define SEQ     2048
#define ROWS    (BATCH * SEQ)              // 4096
#define QKV_ELEMS (BATCH * NHEADS * SEQ * HD)  // 4194304 halves per tensor
#define ATTN_SCALE 0.125f                  // 1/sqrt(64)

// ---- fragment helpers -------------------------------------------------------
// A (16x32 f16): lane L holds row L%16; halves j=0..15 -> K = 16*hi + j
//                (hi = L/16)
static __device__ __forceinline__ v16h ld_afrag(const _Float16* rowp, int hi) {
    v8h lo = *(const v8h*)(rowp + hi * 16);
    v8h hiv = *(const v8h*)(rowp + hi * 16 + 8);
    return __builtin_shufflevector(lo, hiv, 0,1,2,3,4,5,6,7,8,9,10,11,12,13,14,15);
}
// B (32x16 f16): lane L holds col L%16; halves j=0..15 -> K = 16*hi + j
// (contiguous 16-half run starting at p)
static __device__ __forceinline__ v16h ld_bfrag(const _Float16* p) {
    v8h lo = *(const v8h*)p;
    v8h hiv = *(const v8h*)(p + 8);
    return __builtin_shufflevector(lo, hiv, 0,1,2,3,4,5,6,7,8,9,10,11,12,13,14,15);
}
static __device__ __forceinline__ v8f wmma_f16(v16h a, v16h b, v8f c) {
    return __builtin_amdgcn_wmma_f32_16x16x32_f16(false, a, false, b, (short)0, c,
                                                  false, false);
}

// async copy: 16B global -> LDS per lane, tracked by ASYNCcnt (CDNA5)
static __device__ __forceinline__ void async_g2l_b128(void* lds_ptr,
                                                      const void* gptr) {
    unsigned laddr = (unsigned)(unsigned long long)lds_ptr;  // wave-relative LDS addr
    asm volatile("global_load_async_to_lds_b128 %0, %1, off"
                 :: "v"(laddr), "v"(gptr) : "memory");
}
static __device__ __forceinline__ void wait_asynccnt0() {
    asm volatile("s_wait_asynccnt 0x0" ::: "memory");
}

// ============================================================================
// Kernel 1: QKV projection.  C[4096,3072] = x[4096,1024] @ Wqkv + b
// 128x128 block tile, 8 waves (2x4), each wave 64x32 (4x2 WMMA tiles).
// Epilogue: +bias, fold ATTN_SCALE into Q, cast f16, scatter to [B,H,S,hd].
// ============================================================================
__global__ __launch_bounds__(256)
void qkv_gemm_kernel(const float* __restrict__ x,
                     const float* __restrict__ Wqkv,
                     const float* __restrict__ bqkv,
                     _Float16* __restrict__ qbuf,
                     _Float16* __restrict__ kbuf,
                     _Float16* __restrict__ vbuf) {
    __shared__ __align__(16) _Float16 Alds[128 * 32];   // [m][k]
    __shared__ __align__(16) _Float16 Blds[128 * 32];   // [n][k]  (W transposed)

    const int tid  = threadIdx.x;
    const int wave = tid >> 5;
    const int lane = tid & 31;
    const int ln   = lane & 15;
    const int hi   = lane >> 4;
    const int row0 = blockIdx.y * 128;
    const int col0 = blockIdx.x * 128;
    const int wr   = wave >> 2;   // 0..1  -> 64-row slab
    const int wc   = wave & 3;    // 0..3  -> 32-col slab

    v8f acc[4][2];
#pragma unroll
    for (int mt = 0; mt < 4; ++mt)
#pragma unroll
        for (int nt = 0; nt < 2; ++nt) acc[mt][nt] = (v8f)0.0f;

    for (int k0 = 0; k0 < DIM; k0 += 32) {
        // stage A tile 128x32 (f32 -> f16)
#pragma unroll
        for (int i = 0; i < 16; ++i) {
            int e = tid + i * 256;           // 0..4095
            int m = e >> 5, k = e & 31;
            Alds[e] = (_Float16)x[(row0 + m) * DIM + k0 + k];
        }
        // stage B tile 32x128 transposed into [n][k]
#pragma unroll
        for (int i = 0; i < 16; ++i) {
            int e = tid + i * 256;
            int n = e & 127, k = e >> 7;
            Blds[n * 32 + k] = (_Float16)Wqkv[(k0 + k) * (3 * DIM) + col0 + n];
        }
        __syncthreads();

        v16h bfrag[2];
#pragma unroll
        for (int nt = 0; nt < 2; ++nt)
            bfrag[nt] = ld_bfrag(Blds + (wc * 32 + nt * 16 + ln) * 32 + hi * 16);
#pragma unroll
        for (int mt = 0; mt < 4; ++mt) {
            v16h afrag = ld_afrag(Alds + (wr * 64 + mt * 16 + ln) * 32, hi);
#pragma unroll
            for (int nt = 0; nt < 2; ++nt)
                acc[mt][nt] = wmma_f16(afrag, bfrag[nt], acc[mt][nt]);
        }
        __syncthreads();
    }

    // epilogue: scatter into per-head Q/K/V (f16)
#pragma unroll
    for (int nt = 0; nt < 2; ++nt) {
        const int c     = col0 + wc * 32 + nt * 16 + ln;  // global col in [0,3072)
        const int which = c >> 10;
        const int d     = c & 1023;
        const int h     = d >> 6;
        const int hd_i  = d & 63;
        const float bias = bqkv[c];
        const float sc   = (which == 0) ? ATTN_SCALE : 1.0f;
        _Float16* dst = (which == 0) ? qbuf : (which == 1) ? kbuf : vbuf;
#pragma unroll
        for (int mt = 0; mt < 4; ++mt) {
#pragma unroll
            for (int r = 0; r < 8; ++r) {
                int rg = row0 + wr * 64 + mt * 16 + r + 8 * hi;  // [0,4096)
                int bb = rg >> 11, s = rg & 2047;
                dst[(((bb * NHEADS + h) * SEQ) + s) * HD + hd_i] =
                    (_Float16)((acc[mt][nt][r] + bias) * sc);
            }
        }
    }
}

// ============================================================================
// Kernel 2: flash attention per (b,h).  Block = 128 queries (16/wave),
// KV streamed in 32-key tiles.  Double-buffered LDS: K tile arrives via
// global_load_async_to_lds_b128 (ASYNCcnt), V tile prefetched into registers
// and transposed into LDS after compute -> staging overlaps the WMMA work.
// ============================================================================
__global__ __launch_bounds__(256)
void flash_attn_kernel(const _Float16* __restrict__ qbuf,
                       const _Float16* __restrict__ kbuf,
                       const _Float16* __restrict__ vbuf,
                       _Float16* __restrict__ attn) {
    __shared__ __align__(16) _Float16 Klds[2][32 * 64];      // [buf][key][hd]
    __shared__ __align__(16) _Float16 Vtlds[2][64 * 32];     // [buf][hd][key]
    __shared__ __align__(16) _Float16 Plds[8 * 16 * 32];     // per-wave P scratch

    const int tid  = threadIdx.x;
    const int wave = tid >> 5;
    const int lane = tid & 31;
    const int ln   = lane & 15;
    const int hi   = lane >> 4;
    const int bh   = blockIdx.y;            // 0..31 = b*16 + h
    const int h    = bh & 15;
    const int bb   = bh >> 4;
    const int q0   = blockIdx.x * 128 + wave * 16;
    const size_t base = (size_t)bh * SEQ * HD;

    // staging assignment: each thread owns one 16B chunk of the 32x64 tile
    const int skey = tid >> 3;              // 0..31
    const int sc8  = tid & 7;               // 0..7
    const _Float16* kg = kbuf + base + (size_t)skey * HD + sc8 * 8;
    const _Float16* vg = vbuf + base + (size_t)skey * HD + sc8 * 8;

    // Q fragments (16 rows x 64 hd = two 16x32 A-fragments), live whole loop
    v16h qfrag[2];
    {
        const _Float16* qp = qbuf + base + (size_t)(q0 + ln) * HD;
#pragma unroll
        for (int c = 0; c < 2; ++c) {
            v8h lo  = *(const v8h*)(qp + c * 32 + hi * 16);
            v8h hiv = *(const v8h*)(qp + c * 32 + hi * 16 + 8);
            qfrag[c] = __builtin_shufflevector(lo, hiv,
                        0,1,2,3,4,5,6,7,8,9,10,11,12,13,14,15);
        }
    }

    v8f o[4];
#pragma unroll
    for (int t = 0; t < 4; ++t) o[t] = (v8f)0.0f;
    float mrun[8], lrun[8];
#pragma unroll
    for (int r = 0; r < 8; ++r) { mrun[r] = -3.0e38f; lrun[r] = 0.0f; }

    // ---- prologue: stage tile 0 into buffer 0 ----
    {
        async_g2l_b128(&Klds[0][skey * 64 + sc8 * 8], kg);
        v8h vv = *(const v8h*)vg;
#pragma unroll
        for (int i = 0; i < 8; ++i) Vtlds[0][(sc8 * 8 + i) * 32 + skey] = vv[i];
        wait_asynccnt0();
    }
    __syncthreads();

    const int NTILES = SEQ / 32;            // 64
    for (int it = 0; it < NTILES; ++it) {
        const int p = it & 1;
        const bool more = (it + 1 < NTILES);   // uniform across the block
        v8h vnext = {};
        if (more) {
            const size_t koff = (size_t)(it + 1) * 32 * HD;
            async_g2l_b128(&Klds[p ^ 1][skey * 64 + sc8 * 8], kg + koff);
            vnext = *(const v8h*)(vg + koff);   // latency hidden under compute
        }

        // S = Q @ K^T  (16 queries x 32 keys, two C tiles)
        v8f st[2];
#pragma unroll
        for (int kt = 0; kt < 2; ++kt) {
            v8f s = (v8f)0.0f;
#pragma unroll
            for (int c = 0; c < 2; ++c) {
                v16h bf = ld_bfrag(&Klds[p][(kt * 16 + ln) * 64 + c * 32 + hi * 16]);
                s = wmma_f16(qfrag[c], bf, s);
            }
            st[kt] = s;
        }

        // online softmax (row = r + 8*hi, lives in a 16-lane group)
#pragma unroll
        for (int r = 0; r < 8; ++r) {
            float cur = fmaxf(st[0][r], st[1][r]);
            cur = fmaxf(cur, __shfl_xor(cur, 1, 32));
            cur = fmaxf(cur, __shfl_xor(cur, 2, 32));
            cur = fmaxf(cur, __shfl_xor(cur, 4, 32));
            cur = fmaxf(cur, __shfl_xor(cur, 8, 32));
            float mnew  = fmaxf(mrun[r], cur);
            float alpha = __expf(mrun[r] - mnew);
            mrun[r] = mnew;
            float p0 = __expf(st[0][r] - mnew);
            float p1 = __expf(st[1][r] - mnew);
            st[0][r] = p0; st[1][r] = p1;
            float rs = p0 + p1;
            rs += __shfl_xor(rs, 1, 32);
            rs += __shfl_xor(rs, 2, 32);
            rs += __shfl_xor(rs, 4, 32);
            rs += __shfl_xor(rs, 8, 32);
            lrun[r] = lrun[r] * alpha + rs;
#pragma unroll
            for (int t = 0; t < 4; ++t) o[t][r] *= alpha;
        }

        // reshape P (C layout) -> A-fragment layout through wave-private LDS
        _Float16* pw = Plds + wave * 512;
#pragma unroll
        for (int r = 0; r < 8; ++r) {
            int m = r + 8 * hi;
            pw[m * 32 + ln]      = (_Float16)st[0][r];
            pw[m * 32 + 16 + ln] = (_Float16)st[1][r];
        }
        v16h pfrag = ld_afrag(pw + ln * 32, hi);   // wave-local: no barrier

        // O += P @ V   (4 hd tiles of 16)
#pragma unroll
        for (int t = 0; t < 4; ++t) {
            v16h bf = ld_bfrag(&Vtlds[p][(t * 16 + ln) * 32 + hi * 16]);
            o[t] = wmma_f16(pfrag, bf, o[t]);
        }

        // commit next V tile (transposed) into the other buffer
        if (more) {
#pragma unroll
            for (int i = 0; i < 8; ++i)
                Vtlds[p ^ 1][(sc8 * 8 + i) * 32 + skey] = vnext[i];
        }
        wait_asynccnt0();      // K async copy must land before the barrier
        __syncthreads();       // DScnt wait inserted by compiler
    }

    // finalize: O /= l, write [B,S,D] f16
    float inv[8];
#pragma unroll
    for (int r = 0; r < 8; ++r) inv[r] = 1.0f / lrun[r];
#pragma unroll
    for (int t = 0; t < 4; ++t) {
        int dcol = h * 64 + t * 16 + ln;
#pragma unroll
        for (int r = 0; r < 8; ++r) {
            int rg = bb * SEQ + q0 + r + 8 * hi;     // [0,4096)
            attn[(size_t)rg * DIM + dcol] = (_Float16)(o[t][r] * inv[r]);
        }
    }
}

// ============================================================================
// Kernel 3: output projection.  out[4096,1024] = attn @ Wout + b_out  (f32 out)
// ============================================================================
__global__ __launch_bounds__(256)
void out_gemm_kernel(const _Float16* __restrict__ attn,
                     const float* __restrict__ Wout,
                     const float* __restrict__ bout,
                     float* __restrict__ out) {
    __shared__ __align__(16) _Float16 Alds[128 * 32];
    __shared__ __align__(16) _Float16 Blds[128 * 32];

    const int tid  = threadIdx.x;
    const int wave = tid >> 5;
    const int lane = tid & 31;
    const int ln   = lane & 15;
    const int hi   = lane >> 4;
    const int row0 = blockIdx.y * 128;
    const int col0 = blockIdx.x * 128;
    const int wr   = wave >> 2;
    const int wc   = wave & 3;

    v8f acc[4][2];
#pragma unroll
    for (int mt = 0; mt < 4; ++mt)
#pragma unroll
        for (int nt = 0; nt < 2; ++nt) acc[mt][nt] = (v8f)0.0f;

    for (int k0 = 0; k0 < DIM; k0 += 32) {
        // stage A (already f16): 512 16B chunks, 2 per thread
#pragma unroll
        for (int i = 0; i < 2; ++i) {
            int cidx = tid + i * 256;
            int m = cidx >> 2, kb = cidx & 3;
            *(v8h*)&Alds[m * 32 + kb * 8] =
                *(const v8h*)&attn[(size_t)(row0 + m) * DIM + k0 + kb * 8];
        }
        // stage B transposed [n][k]
#pragma unroll
        for (int i = 0; i < 16; ++i) {
            int e = tid + i * 256;
            int n = e & 127, k = e >> 7;
            Blds[n * 32 + k] = (_Float16)Wout[(k0 + k) * DIM + col0 + n];
        }
        __syncthreads();

        v16h bfrag[2];
#pragma unroll
        for (int nt = 0; nt < 2; ++nt)
            bfrag[nt] = ld_bfrag(Blds + (wc * 32 + nt * 16 + ln) * 32 + hi * 16);
#pragma unroll
        for (int mt = 0; mt < 4; ++mt) {
            v16h afrag = ld_afrag(Alds + (wr * 64 + mt * 16 + ln) * 32, hi);
#pragma unroll
            for (int nt = 0; nt < 2; ++nt)
                acc[mt][nt] = wmma_f16(afrag, bfrag[nt], acc[mt][nt]);
        }
        __syncthreads();
    }

#pragma unroll
    for (int nt = 0; nt < 2; ++nt) {
        const int c = col0 + wc * 32 + nt * 16 + ln;
        const float bias = bout[c];
#pragma unroll
        for (int mt = 0; mt < 4; ++mt) {
#pragma unroll
            for (int r = 0; r < 8; ++r) {
                int rg = row0 + wr * 64 + mt * 16 + r + 8 * hi;
                out[(size_t)rg * DIM + c] = acc[mt][nt][r] + bias;
            }
        }
    }
}

// ============================================================================
extern "C" void kernel_launch(void* const* d_in, const int* in_sizes, int n_in,
                              void* d_out, int out_size, void* d_ws, size_t ws_size,
                              hipStream_t stream) {
    const float* x    = (const float*)d_in[0];
    const float* Wqkv = (const float*)d_in[1];
    const float* bqkv = (const float*)d_in[2];
    const float* Wout = (const float*)d_in[3];
    const float* bout = (const float*)d_in[4];
    float* out = (float*)d_out;

    // workspace: Q | K | V | attn-out, all f16 (32 MB total)
    _Float16* qbuf = (_Float16*)d_ws;
    _Float16* kbuf = qbuf + QKV_ELEMS;
    _Float16* vbuf = kbuf + QKV_ELEMS;
    _Float16* attn = vbuf + QKV_ELEMS;

    qkv_gemm_kernel<<<dim3(3 * DIM / 128, ROWS / 128), 256, 0, stream>>>(
        x, Wqkv, bqkv, qbuf, kbuf, vbuf);
    flash_attn_kernel<<<dim3(SEQ / 128, BATCH * NHEADS), 256, 0, stream>>>(
        qbuf, kbuf, vbuf, attn);
    out_gemm_kernel<<<dim3(DIM / 128, ROWS / 128), 256, 0, stream>>>(
        attn, Wout, bout, out);
}
